// Self_Attn2D_128849019181
// MI455X (gfx1250) — matrix operand order repaired and checked by reading, not compiled, not run on hardware
//
#include <hip/hip_runtime.h>
#include <hip/hip_bf16.h>

// ---------------------------------------------------------------------------
// Self-attention 2D for MI455X (gfx1250, wave32, WMMA + TDM).
// B=4, C=256, W=H=64 -> N=4096, CQK=32.
// Outputs: out (B*C*N f32) ++ attention (B*N*N f32), concatenated in d_out.
// ---------------------------------------------------------------------------

#define BB   4
#define CC   256
#define NN   4096
#define CQK  32

typedef __attribute__((ext_vector_type(16))) _Float16     v16h;
typedef __attribute__((ext_vector_type(8)))  float        v8f;
typedef __attribute__((ext_vector_type(4)))  unsigned int v4u;
typedef __attribute__((ext_vector_type(8)))  int          v8i;
typedef __attribute__((ext_vector_type(4)))  int          v4i;

// ---- WMMA operand builders (contiguous k order, same in A and B, wave32) ----------

// A matrix 16x32 f16, element (m,k) = tile[m*ld + k] (row-major in memory).
// lanes 0-15: m=lane, K=0..15 ; lanes 16-31: m=lane-16, K=16..31.
__device__ __forceinline__ v16h load_a_rowmajor_f16(const _Float16* tile, int ld, int lane) {
    const int m    = lane & 15;
    const int koff = (lane >> 4) * 16;
    const _Float16* p = tile + (size_t)m * ld;
    v16h a;
#pragma unroll
    for (int r = 0; r < 8; ++r) {
        const int k = koff + 2 * r;
        a[2 * r]     = p[k];
        a[2 * r + 1] = p[k + 1];
    }
    return a;
}

// B matrix 32x16 f16, element (k,n) = tileT[n*ld + k] ("n-major" memory).
// lanes 0-15: n=lane, K=0..15 ; lanes 16-31: n=lane-16, K=16..31.
__device__ __forceinline__ v16h load_b_nmajor_f16(const _Float16* tileT, int ld, int lane) {
    const int n    = lane & 15;
    const int koff = (lane >> 4) * 16;
    const _Float16* p = tileT + (size_t)n * ld + koff;
    v16h b;
#pragma unroll
    for (int r = 0; r < 16; ++r) b[r] = p[r];
    return b;
}

// ---- TDM: DMA one 16-row x 32-col f32 tile (row stride NN) into LDS --------
// D# per CDNA5 ISA sec 8: group0 = control/lds/global addr, group1 = dims.
// LDS padding: +1 DWORD after every 32 DWORDs -> rows land at stride 33 floats
// (bank-conflict-free), matching the at[16][33] consumer layout.
__device__ __forceinline__ void tdm_load_attn_tile(const float* gptr, void* ldsptr) {
    const unsigned long long ga  = (unsigned long long)(size_t)gptr;
    const unsigned int       lds = (unsigned int)(size_t)ldsptr;

    v4u g0;
    g0[0] = 1u;                                    // count=1, user descriptor
    g0[1] = lds;                                   // lds_addr (byte)
    g0[2] = (unsigned int)ga;                      // global_addr[31:0]
    g0[3] = (unsigned int)(ga >> 32) | (2u << 30); // global_addr[56:32] | type=2

    v8i g1;
    g1[0] = (2 << 16)      // data_size = 4 bytes
          | (1 << 20)      // pad_enable
          | (4 << 22)      // pad_interval: 32 DWORDs
          | (0 << 25);     // pad_amount: 1 DWORD
    g1[1] = (int)((unsigned)NN << 16);   // tensor_dim0[15:0] = 4096 (hi=0)
    g1[2] = (int)((unsigned)NN << 16);   // dim0 hi16=0 | tensor_dim1[15:0] = 4096
    g1[3] = (32 << 16);                  // dim1 hi16=0 | tile_dim0 = 32
    g1[4] = 16;                          // tile_dim1 = 16, tile_dim2 = 0
    g1[5] = NN;                          // tensor_dim0_stride[31:0] = 4096
    g1[6] = 0;                           // stride0 hi | tensor_dim1_stride lo
    g1[7] = 0;

    v4i g2 = {0, 0, 0, 0};
    v4i g3 = {0, 0, 0, 0};

#if __clang_major__ >= 23
    v8i g4 = {0, 0, 0, 0, 0, 0, 0, 0};
    __builtin_amdgcn_tensor_load_to_lds(g0, g1, g2, g3, g4, 0);
#else
    __builtin_amdgcn_tensor_load_to_lds(g0, g1, g2, g3, 0);
#endif
}

// ---- Kernel 1: Q and K projections (x read exactly once) -------------------
__global__ void qk_kernel(const float* __restrict__ x,
                          const float* __restrict__ wq, const float* __restrict__ bq,
                          const float* __restrict__ wk, const float* __restrict__ bk,
                          _Float16* __restrict__ qws, _Float16* __restrict__ kws) {
    const int idx = blockIdx.x * blockDim.x + threadIdx.x;   // B*N threads
    const int n = idx & (NN - 1);
    const int b = idx >> 12;

    float aq[CQK], ak[CQK];
#pragma unroll
    for (int o = 0; o < CQK; ++o) { aq[o] = 0.f; ak[o] = 0.f; }

    const float* xb = x + (size_t)b * CC * NN + n;
    for (int c = 0; c < CC; ++c) {
        const float xv = xb[(size_t)c * NN];
#pragma unroll
        for (int o = 0; o < CQK; ++o) {
            aq[o] = fmaf(wq[o * CC + c], xv, aq[o]);
            ak[o] = fmaf(wk[o * CC + c], xv, ak[o]);
        }
    }
    _Float16* qr = qws + ((size_t)b * NN + n) * CQK;
    _Float16* kr = kws + ((size_t)b * NN + n) * CQK;
#pragma unroll
    for (int o = 0; o < CQK; ++o) {
        qr[o] = (_Float16)(aq[o] + bq[o]);
        kr[o] = (_Float16)(ak[o] + bk[o]);
    }
}

// ---- Kernel 2: V projection ------------------------------------------------
__global__ void v_kernel(const float* __restrict__ x,
                         const float* __restrict__ wv, const float* __restrict__ bv,
                         _Float16* __restrict__ vws) {
    const int idx   = blockIdx.x * blockDim.x + threadIdx.x;  // B*(C/16)*N threads
    const int n     = idx & (NN - 1);
    const int chunk = (idx >> 12) & 15;
    const int b     = idx >> 16;

    float acc[16];
#pragma unroll
    for (int t = 0; t < 16; ++t) acc[t] = 0.f;

    const float* xb = x + (size_t)b * CC * NN + n;
    const int o0 = chunk * 16;
    for (int c = 0; c < CC; ++c) {
        const float xv = xb[(size_t)c * NN];
#pragma unroll
        for (int t = 0; t < 16; ++t)
            acc[t] = fmaf(wv[(o0 + t) * CC + c], xv, acc[t]);
    }
#pragma unroll
    for (int t = 0; t < 16; ++t)
        vws[((size_t)b * CC + o0 + t) * NN + n] = (_Float16)(acc[t] + bv[o0 + t]);
}

// ---- Kernel 3: fused energy (WMMA) + row softmax ---------------------------
// Block = (i-tile of 16 rows, batch). 8 waves x 32 WMMA tiles; raw energy
// lives only in 256 KB of WGP LDS, softmax'd rows stream to HBM exactly once.
#define LDE 4100
__global__ void energy_softmax_kernel(const _Float16* __restrict__ qws,
                                      const _Float16* __restrict__ kws,
                                      float* __restrict__ attn) {
    extern __shared__ float sm[];                 // [16][LDE]
    const int b    = blockIdx.y;
    const int i0   = blockIdx.x * 16;
    const int wave = threadIdx.x >> 5;
    const int lane = threadIdx.x & 31;

    const _Float16* qtile = qws + ((size_t)b * NN + i0) * CQK;
    const v16h a = load_a_rowmajor_f16(qtile, CQK, lane);

    const int mrow = (lane >> 4) * 8;
    const int nn   = lane & 15;
#pragma unroll 1
    for (int t = 0; t < 32; ++t) {
        const int j0 = (wave * 32 + t) * 16;
        const _Float16* ktile = kws + ((size_t)b * NN + j0) * CQK;
        const v16h bmat = load_b_nmajor_f16(ktile, CQK, lane);
        v8f c = {};
        c = __builtin_amdgcn_wmma_f32_16x16x32_f16(
                false, a, false, bmat, (short)0, c, false, false);
#pragma unroll
        for (int r = 0; r < 8; ++r)
            sm[(size_t)(mrow + r) * LDE + j0 + nn] = c[r];
    }
    __syncthreads();

    // Softmax: each wave owns 2 of the 16 rows.
#pragma unroll 1
    for (int rr = 0; rr < 2; ++rr) {
        const int row = wave * 2 + rr;
        float* rp = sm + (size_t)row * LDE;

        float mx = -3.402823466e+38f;
#pragma unroll 4
        for (int tt = 0; tt < NN / 32; ++tt)
            mx = fmaxf(mx, rp[lane + 32 * tt]);
#pragma unroll
        for (int off = 16; off > 0; off >>= 1)
            mx = fmaxf(mx, __shfl_xor(mx, off, 32));

        float s = 0.f;
#pragma unroll 4
        for (int tt = 0; tt < NN / 32; ++tt) {
            const float e = __expf(rp[lane + 32 * tt] - mx);
            rp[lane + 32 * tt] = e;
            s += e;
        }
#pragma unroll
        for (int off = 16; off > 0; off >>= 1)
            s += __shfl_xor(s, off, 32);
        const float inv = 1.0f / s;

        float* arow = attn + ((size_t)b * NN + i0 + row) * NN;
#pragma unroll 4
        for (int tt = 0; tt < NN / 32; ++tt)
            arow[lane + 32 * tt] = rp[lane + 32 * tt] * inv;
    }
}

// ---- Kernel 4: out = gamma * (V x attn^T) + x  (WMMA, TDM-staged attn) -----
// Block = (i-tile of 16 cols, batch). The 16x32 attention block is DMA'd into
// LDS by the Tensor Data Mover each K-step and shared by all 8 waves; each
// wave owns two 16-channel tiles (2 WMMA per step, 128 steps).
__global__ void out_kernel(const _Float16* __restrict__ vws,
                           const float* __restrict__ attn,
                           const float* __restrict__ x,
                           const float* __restrict__ gamma,
                           float* __restrict__ out) {
    __shared__ __align__(16) float at[16][33];    // TDM pads rows to stride 33
    const int b    = blockIdx.y;
    const int i0   = blockIdx.x * 16;
    const int wave = threadIdx.x >> 5;
    const int lane = threadIdx.x & 31;
    const int co0  = wave * 32;

    v8f acc0 = {}, acc1 = {};
    const int n  = lane & 15;
    const int kh = (lane >> 4) * 16;
    const float* atile0 = attn + ((size_t)b * NN + i0) * NN;

#pragma unroll 1
    for (int js = 0; js < NN / 32; ++js) {
        const int j0 = js * 32;
        __syncthreads();                          // LDS tile free to overwrite
        if (wave == 0) {
            tdm_load_attn_tile(atile0 + j0, (void*)&at[0][0]);
            __builtin_amdgcn_s_wait_tensorcnt(0);
        } else if (wave == 1 && js + 1 < NN / 32) {
            // warm L2 for the next TDM tile (one row per lane)
            __builtin_prefetch(atile0 + (size_t)(lane & 15) * NN + j0 + 32, 0, 0);
        }
        __syncthreads();                          // tile visible to all waves

        // B operand (32x16): element (k=j_local, n=i_local) = at[n][k]
        v16h bmat;
#pragma unroll
        for (int r = 0; r < 16; ++r) bmat[r] = (_Float16)at[n][kh + r];

        const _Float16* v0 = vws + ((size_t)b * CC + co0) * NN + j0;
        const _Float16* v1 = vws + ((size_t)b * CC + co0 + 16) * NN + j0;
        const v16h a0 = load_a_rowmajor_f16(v0, NN, lane);
        const v16h a1 = load_a_rowmajor_f16(v1, NN, lane);

        acc0 = __builtin_amdgcn_wmma_f32_16x16x32_f16(
                   false, a0, false, bmat, (short)0, acc0, false, false);
        acc1 = __builtin_amdgcn_wmma_f32_16x16x32_f16(
                   false, a1, false, bmat, (short)0, acc1, false, false);
    }

    const float g  = gamma[0];
    const int mb   = (lane >> 4) * 8;
    const int icol = i0 + (lane & 15);
#pragma unroll
    for (int r = 0; r < 8; ++r) {
        {
            const int co = co0 + mb + r;
            const size_t o = ((size_t)b * CC + co) * NN + icol;
            out[o] = fmaf(g, acc0[r], x[o]);
        }
        {
            const int co = co0 + 16 + mb + r;
            const size_t o = ((size_t)b * CC + co) * NN + icol;
            out[o] = fmaf(g, acc1[r], x[o]);
        }
    }
}

// ---------------------------------------------------------------------------
extern "C" void kernel_launch(void* const* d_in, const int* in_sizes, int n_in,
                              void* d_out, int out_size, void* d_ws, size_t ws_size,
                              hipStream_t stream) {
    const float* x     = (const float*)d_in[0];
    const float* wq    = (const float*)d_in[1];
    const float* bq    = (const float*)d_in[2];
    const float* wk    = (const float*)d_in[3];
    const float* bk    = (const float*)d_in[4];
    const float* wv    = (const float*)d_in[5];
    const float* bv    = (const float*)d_in[6];
    const float* gamma = (const float*)d_in[7];

    float* out  = (float*)d_out;
    float* attn = out + (size_t)BB * CC * NN;     // 2nd tuple element

    _Float16* qws = (_Float16*)d_ws;                           // B*N*32 f16
    _Float16* kws = qws + (size_t)BB * NN * CQK;               // B*N*32 f16
    _Float16* vws = kws + (size_t)BB * NN * CQK;               // B*C*N  f16

    qk_kernel<<<(BB * NN) / 256, 256, 0, stream>>>(x, wq, bq, wk, bk, qws, kws);

    v_kernel<<<(BB * (CC / 16) * NN) / 256, 256, 0, stream>>>(x, wv, bv, vws);

    const size_t esm_bytes = (size_t)16 * LDE * sizeof(float); // ~256 KB LDS
    energy_softmax_kernel<<<dim3(NN / 16, BB), 256, esm_bytes, stream>>>(
        qws, kws, attn);

    out_kernel<<<dim3(NN / 16, BB), 256, 0, stream>>>(vws, attn, x, gamma, out);
}
